// SimpleBNN_60455959658594
// MI455X (gfx1250) — 16-bit and 8-bit matrix operand order repaired and checked by reading, compiled, not run on hardware
//
#include <hip/hip_runtime.h>
#include <stdint.h>

typedef __attribute__((ext_vector_type(8)))  int          v8i;
typedef __attribute__((ext_vector_type(4)))  int          v4i;
typedef __attribute__((ext_vector_type(4)))  unsigned int v4u;
typedef __attribute__((ext_vector_type(8)))  float        v8f;
typedef __attribute__((ext_vector_type(4)))  float        v4f;
typedef __attribute__((ext_vector_type(16))) _Float16     v16h;

#define TM 128
#define TN 128
#define TK 128
#define BP 144
#define TILEB (TM * BP)
#define BUFB  (2 * TILEB)

#define FP 80
#define FTILEB (TM * FP)
#define FBUFB  (2 * FTILEB)

__device__ __forceinline__ int8_t fsign(float v) {
  return v > 0.f ? (int8_t)1 : (v < 0.f ? (int8_t)-1 : (int8_t)0);
}

__global__ __launch_bounds__(256) void binarize_kernel(const float* __restrict__ in,
                                                       int8_t* __restrict__ out, int n4) {
  int i = blockIdx.x * 256 + threadIdx.x;
  if (i >= n4) return;
  v4f x = *(const v4f*)(in + (size_t)i * 4);
  union { int w; int8_t b[4]; } u;
  u.b[0] = fsign(x[0]); u.b[1] = fsign(x[1]); u.b[2] = fsign(x[2]); u.b[3] = fsign(x[3]);
  *(int*)(out + (size_t)i * 4) = u.w;
}

__device__ __forceinline__ void tdm_load_2d(const void* gbase, unsigned lds_addr,
                                            unsigned tile_d0, unsigned tile_d1,
                                            unsigned tdim0, unsigned tdim1,
                                            unsigned long long row_stride,
                                            unsigned pad_iv, unsigned pad_amt) {
  unsigned long long ga = (unsigned long long)(uintptr_t)gbase;
  v4u g0;
  g0[0] = 1u;
  g0[1] = lds_addr;
  g0[2] = (unsigned)(ga & 0xFFFFFFFFull);
  g0[3] = (unsigned)((ga >> 32) & 0x1FFFFFFull) | (2u << 30);
  v8i g1;
  g1[0] = (int)((1u << 20) | (pad_iv << 22) | (pad_amt << 25));
  g1[1] = (int)((tdim0 & 0xFFFFu) << 16);
  g1[2] = (int)(((tdim0 >> 16) & 0xFFFFu) | ((tdim1 & 0xFFFFu) << 16));
  g1[3] = (int)(((tdim1 >> 16) & 0xFFFFu) | ((tile_d0 & 0xFFFFu) << 16));
  g1[4] = (int)(tile_d1 & 0xFFFFu);
  g1[5] = (int)(unsigned)(row_stride & 0xFFFFFFFFull);
  g1[6] = (int)(unsigned)((row_stride >> 32) & 0xFFFFull);
  g1[7] = 0;
  v4i g2 = {0, 0, 0, 0}, g3 = {0, 0, 0, 0};
  v8i g4 = {0, 0, 0, 0, 0, 0, 0, 0};
  __builtin_amdgcn_tensor_load_to_lds(g0, g1, g2, g3, g4, 0);
}

__device__ __forceinline__ v8i ldsA_frag_i8(const char* tbase, int rowBase, int kh, int lane) {
  int r = rowBase + (lane & 15);
  const char* p = tbase + r * BP + kh * 64 + ((lane >> 4) << 4);
  union { v8i v; v4i q[2]; } u;
  u.q[0] = *(const v4i*)(p);
  u.q[1] = *(const v4i*)(p + 32);
  return u.v;
}
__device__ __forceinline__ v8i ldsB_frag_i8(const char* tbase, int colBase, int kh, int lane) {
  int c = colBase + (lane & 15);
  const char* p = tbase + c * BP + kh * 64 + ((lane >> 4) << 4);
  union { v8i v; v4i q[2]; } u;
  u.q[0] = *(const v4i*)(p);
  u.q[1] = *(const v4i*)(p + 32);
  return u.v;
}
__device__ __forceinline__ v16h ldsA_frag_f16(const char* tbase, int rowBase, int lane) {
  int r = rowBase + (lane & 15);
  const char* p = tbase + r * FP + ((lane >> 4) << 4);
  union { v16h v; v4i q[2]; } u;
  u.q[0] = *(const v4i*)(p);
  u.q[1] = *(const v4i*)(p + 32);
  return u.v;
}
__device__ __forceinline__ v16h ldsB_frag_f16(const char* tbase, int colBase, int lane) {
  int c = colBase + (lane & 15);
  const char* p = tbase + c * FP + ((lane >> 4) << 4);
  union { v16h v; v4i q[2]; } u;
  u.q[0] = *(const v4i*)(p);
  u.q[1] = *(const v4i*)(p + 32);
  return u.v;
}

__global__ __launch_bounds__(256)
void bgemm_sign_kernel(const int8_t* __restrict__ A, const int8_t* __restrict__ W,
                       const float* __restrict__ bias, int8_t* __restrict__ out,
                       int M, int N, int K) {
  __shared__ __align__(16) char smem[2 * BUFB];
  const int tid   = threadIdx.x;
  const int lane  = tid & 31;
  const int wave  = tid >> 5;
  const int waveM = wave >> 2;
  const int waveN = wave & 3;
  const int mBase = blockIdx.x * TM;
  const int nBase = blockIdx.y * TN;
  const int KT = K / TK;
  const unsigned lds0 = (unsigned)(uintptr_t)&smem[0];

  const v8i zero8 = {0, 0, 0, 0, 0, 0, 0, 0};
  v8i acc[4][2];
#pragma unroll
  for (int mi = 0; mi < 4; ++mi)
#pragma unroll
    for (int ni = 0; ni < 2; ++ni) acc[mi][ni] = zero8;

  if (wave == 0)
    tdm_load_2d(A + (size_t)mBase * K, lds0, TK, TM, (unsigned)K, (unsigned)M,
                (unsigned long long)K, 4u, 3u);
  else if (wave == 1)
    tdm_load_2d(W + (size_t)nBase * K, lds0 + TILEB, TK, TN, (unsigned)K, (unsigned)N,
                (unsigned long long)K, 4u, 3u);
  if (wave < 2) __builtin_amdgcn_s_wait_tensorcnt(0);
  __syncthreads();

  for (int kt = 0; kt < KT; ++kt) {
    const char* sA = smem + (kt & 1) * BUFB;
    const char* sB = sA + TILEB;

    if (kt + 1 < KT) {
      const unsigned dst = lds0 + ((kt + 1) & 1) * BUFB;
      const size_t k0 = (size_t)(kt + 1) * TK;
      if (wave == 0)
        tdm_load_2d(A + (size_t)mBase * K + k0, dst, TK, TM, (unsigned)K, (unsigned)M,
                    (unsigned long long)K, 4u, 3u);
      else if (wave == 1)
        tdm_load_2d(W + (size_t)nBase * K + k0, dst + TILEB, TK, TN, (unsigned)K, (unsigned)N,
                    (unsigned long long)K, 4u, 3u);
    }

#pragma unroll
    for (int kh = 0; kh < 2; ++kh) {
      v8i af[4], bf[2];
#pragma unroll
      for (int mi = 0; mi < 4; ++mi) af[mi] = ldsA_frag_i8(sA, waveM * 64 + mi * 16, kh, lane);
#pragma unroll
      for (int ni = 0; ni < 2; ++ni) bf[ni] = ldsB_frag_i8(sB, waveN * 32 + ni * 16, kh, lane);
#pragma unroll
      for (int mi = 0; mi < 4; ++mi)
#pragma unroll
        for (int ni = 0; ni < 2; ++ni)
          acc[mi][ni] = __builtin_amdgcn_wmma_i32_16x16x64_iu8(
              true, af[mi], true, bf[ni], acc[mi][ni], false, false);
    }

    if (wave < 2 && kt + 1 < KT) __builtin_amdgcn_s_wait_tensorcnt(0);
    __syncthreads();
  }

#pragma unroll
  for (int mi = 0; mi < 4; ++mi) {
#pragma unroll
    for (int ni = 0; ni < 2; ++ni) {
      const int n = nBase + waveN * 32 + ni * 16 + (lane & 15);
      const float bv = bias[n];
#pragma unroll
      for (int j = 0; j < 8; ++j) {
        const int m = mBase + waveM * 64 + mi * 16 + (lane >> 4) * 8 + j;
        out[(size_t)m * N + n] = fsign((float)acc[mi][ni][j] + bv);
      }
    }
  }
}

__global__ __launch_bounds__(256)
void final_gemm_kernel(const int8_t* __restrict__ A, const float* __restrict__ W,
                       const float* __restrict__ bias, float* __restrict__ out,
                       int M, int N, int K) {
  __shared__ __align__(16) char smem[2 * FBUFB];
  const int tid   = threadIdx.x;
  const int lane  = tid & 31;
  const int wave  = tid >> 5;
  const int waveM = wave >> 2;
  const int waveN = wave & 3;
  const int mBase = blockIdx.x * TM;
  const int nBase = blockIdx.y * TN;
  const int KT = K >> 5;

  const int c0 = tid, c1 = tid + 256;
  const int ar0 = c0 >> 2, as0 = c0 & 3;
  const int ar1 = c1 >> 2, as1 = c1 & 3;

  const v8f zero8 = {0.f, 0.f, 0.f, 0.f, 0.f, 0.f, 0.f, 0.f};
  v8f acc[4][2];
#pragma unroll
  for (int mi = 0; mi < 4; ++mi)
#pragma unroll
    for (int ni = 0; ni < 2; ++ni) acc[mi][ni] = zero8;

  auto cvtA = [](long long v) -> v4i {
    union { v4i q; _Float16 h[8]; } u;
#pragma unroll
    for (int i = 0; i < 8; ++i) u.h[i] = (_Float16)(float)(int)(int8_t)(v >> (8 * i));
    return u.q;
  };
  auto cvtB = [](v4f lo, v4f hi) -> v4i {
    union { v4i q; _Float16 h[8]; } u;
#pragma unroll
    for (int i = 0; i < 4; ++i) { u.h[i] = (_Float16)lo[i]; u.h[4 + i] = (_Float16)hi[i]; }
    return u.q;
  };
  const v4f zf = {0.f, 0.f, 0.f, 0.f};

  auto stage = [&](int kt, char* dst) {
    const int k0 = kt << 5;
    char* dA = dst;  char* dB = dst + FTILEB;
    long long a0 = *(const long long*)(A + (size_t)(mBase + ar0) * K + k0 + as0 * 8);
    long long a1 = *(const long long*)(A + (size_t)(mBase + ar1) * K + k0 + as1 * 8);
    v4f b0l = zf, b0h = zf, b1l = zf, b1h = zf;
    if (nBase + ar0 < N) {
      const float* p = W + (size_t)(nBase + ar0) * K + k0 + as0 * 8;
      b0l = *(const v4f*)(p); b0h = *(const v4f*)(p + 4);
    }
    if (nBase + ar1 < N) {
      const float* p = W + (size_t)(nBase + ar1) * K + k0 + as1 * 8;
      b1l = *(const v4f*)(p); b1h = *(const v4f*)(p + 4);
    }
    *(v4i*)(dA + ar0 * FP + as0 * 16) = cvtA(a0);
    *(v4i*)(dA + ar1 * FP + as1 * 16) = cvtA(a1);
    *(v4i*)(dB + ar0 * FP + as0 * 16) = cvtB(b0l, b0h);
    *(v4i*)(dB + ar1 * FP + as1 * 16) = cvtB(b1l, b1h);
  };

  stage(0, smem);
  __syncthreads();

  for (int kt = 0; kt < KT; ++kt) {
    const char* sA = smem + (kt & 1) * FBUFB;
    const char* sB = sA + FTILEB;

    v16h af[4], bf[2];
#pragma unroll
    for (int mi = 0; mi < 4; ++mi) af[mi] = ldsA_frag_f16(sA, waveM * 64 + mi * 16, lane);
#pragma unroll
    for (int ni = 0; ni < 2; ++ni) bf[ni] = ldsB_frag_f16(sB, waveN * 32 + ni * 16, lane);
#pragma unroll
    for (int mi = 0; mi < 4; ++mi)
#pragma unroll
      for (int ni = 0; ni < 2; ++ni)
        acc[mi][ni] = __builtin_amdgcn_wmma_f32_16x16x32_f16(
            false, af[mi], false, bf[ni], (short)0, acc[mi][ni], false, false);

    const int ktn = (kt + 1 < KT) ? (kt + 1) : kt;
    stage(ktn, smem + ((kt + 1) & 1) * FBUFB);
    __syncthreads();
  }

#pragma unroll
  for (int mi = 0; mi < 4; ++mi) {
#pragma unroll
    for (int ni = 0; ni < 2; ++ni) {
      const int n = nBase + waveN * 32 + ni * 16 + (lane & 15);
      if (n < N) {
        const float bv = bias[n];
#pragma unroll
        for (int j = 0; j < 8; ++j) {
          const int m = mBase + waveM * 64 + mi * 16 + (lane >> 4) * 8 + j;
          out[(size_t)m * N + n] = acc[mi][ni][j] + bv;
        }
      }
    }
  }
}

extern "C" void kernel_launch(void* const* d_in, const int* in_sizes, int n_in,
                              void* d_out, int out_size, void* d_ws, size_t ws_size,
                              hipStream_t stream) {
  constexpr int Bz = 8192, D = 4096, H = 4096, C = 1000;
  const float* x  = (const float*)d_in[0];
  const float* w1 = (const float*)d_in[1];
  const float* b1 = (const float*)d_in[2];
  const float* w2 = (const float*)d_in[3];
  const float* b2 = (const float*)d_in[4];
  const float* w3 = (const float*)d_in[5];
  const float* b3 = (const float*)d_in[6];
  float* out = (float*)d_out;

  char* ws = (char*)d_ws;
  const size_t MB = 1024ull * 1024ull;
  int8_t* xbin  = (int8_t*)(ws);
  int8_t* w1bin = (int8_t*)(ws + 32 * MB);
  int8_t* w2bin = (int8_t*)(ws + 48 * MB);
  int8_t* h1bin = (int8_t*)(ws + 64 * MB);
  int8_t* h2bin = xbin;

  binarize_kernel<<<(Bz * D / 4 + 255) / 256, 256, 0, stream>>>(x, xbin, Bz * D / 4);
  binarize_kernel<<<(H * D / 4 + 255) / 256, 256, 0, stream>>>(w1, w1bin, H * D / 4);
  binarize_kernel<<<(H * H / 4 + 255) / 256, 256, 0, stream>>>(w2, w2bin, H * H / 4);

  bgemm_sign_kernel<<<dim3(Bz / TM, H / TN), 256, 0, stream>>>(xbin, w1bin, b1, h1bin, Bz, H, D);
  bgemm_sign_kernel<<<dim3(Bz / TM, H / TN), 256, 0, stream>>>(h1bin, w2bin, b2, h2bin, Bz, H, H);

  final_gemm_kernel<<<dim3(Bz / TM, (C + TN - 1) / TN), 256, 0, stream>>>(h2bin, w3, b3, out, Bz, C, H);
}
